// Head_22763326669452
// MI455X (gfx1250) — compile-verified, matrix operand order checked by reading, not run on hardware
//
#include <hip/hip_runtime.h>

typedef _Float16 f16;
typedef __attribute__((ext_vector_type(16))) _Float16 v16h;
typedef __attribute__((ext_vector_type(8)))  _Float16 v8h;
typedef __attribute__((ext_vector_type(8)))  float    v8f;
typedef __attribute__((ext_vector_type(4)))  float    v4f;
typedef __attribute__((ext_vector_type(4)))  unsigned int u32x4;
typedef __attribute__((ext_vector_type(8)))  int      i32x8;
typedef __attribute__((ext_vector_type(4)))  int      i32x4;

#define T_DIM 2048
#define B_DIM 8
#define C_DIM 1024
#define H_DIM 128
#define ROWS (T_DIM * B_DIM)   // 16384 (t-major, b inner: row r -> t=r/B, b=r%B)

static __device__ __forceinline__ v8f wmma16(v16h a, v16h b, v8f c) {
    // D(16x16,f32) = A(16x32,f16) * B(32x16,f16) + C(16x16,f32)
    return __builtin_amdgcn_wmma_f32_16x16x32_f16(
        /*neg_a=*/false, a, /*neg_b=*/false, b,
        /*c_mod=*/(short)0, c, /*reuse_a=*/false, /*reuse_b=*/false);
}

static __device__ __forceinline__ v16h cat8(v8h lo, v8h hi) {
    return __builtin_shufflevector(lo, hi, 0, 1, 2, 3, 4, 5, 6, 7,
                                           8, 9, 10, 11, 12, 13, 14, 15);
}

// ---------------------------------------------------------------------------
// TDM 2-D tile load (f16 elements): tile_h rows x tile_w elems, row-major
// compaction into LDS at lds_off.  D# per CDNA5 ISA.
// Tensor dims == tile dims (tile fully in bounds).  Tracked by TENSORcnt.
// This toolchain's builtin is the 6-arg variant (extra i32x8 group, zeroed).
// ---------------------------------------------------------------------------
static __device__ __forceinline__ void tdm_load_2d_f16(
    unsigned lds_off, const f16* gptr,
    unsigned tile_w, unsigned tile_h, unsigned row_stride_elems) {
    const unsigned long long ga = (unsigned long long)gptr;
    u32x4 g0;
    g0.x = 1u;                                               // count=1, user D#
    g0.y = lds_off;                                          // lds_addr (bytes)
    g0.z = (unsigned)ga;                                     // global_addr[31:0]
    g0.w = ((unsigned)(ga >> 32) & 0x01FFFFFFu) | 0x80000000u; // [56:32] | type=2
    i32x8 g1;
    g1[0] = (int)(1u << 16);                                 // data_size=1 (2B)
    g1[1] = (int)((tile_w & 0xFFFFu) << 16);                 // tensor_dim0 lo16
    g1[2] = (int)((tile_w >> 16) | ((tile_h & 0xFFFFu) << 16)); // d0 hi | d1 lo
    g1[3] = (int)((tile_h >> 16) | ((tile_w & 0xFFFFu) << 16)); // d1 hi | tile_dim0
    g1[4] = (int)(tile_h & 0xFFFFu);                         // tile_dim1 (tile_dim2=0)
    g1[5] = (int)row_stride_elems;                           // tensor_dim0_stride lo32
    g1[6] = 0;                                               // stride hi | d1_stride lo
    g1[7] = 0;
    const i32x4 zero4 = {0, 0, 0, 0};
    const i32x8 zero8 = {0, 0, 0, 0, 0, 0, 0, 0};
    __builtin_amdgcn_tensor_load_to_lds(g0, g1, zero4, zero4, zero8, 0);
}

// ---------------------------------------------------------------------------
// Weight prep: Wt[h][c] = (f16) W[c][h]   (contraction axis made contiguous)
// ---------------------------------------------------------------------------
__global__ __launch_bounds__(256)
void wtrans_kernel(const float* __restrict__ W, f16* __restrict__ Wt) {
    const int idx = blockIdx.x * 256 + threadIdx.x;   // over C*H = 131072
    const int c = idx >> 7;          // / H_DIM
    const int h = idx & (H_DIM - 1);
    Wt[(size_t)h * C_DIM + c] = (f16)W[idx];
}

// ---------------------------------------------------------------------------
// Projection GEMM (f16 in / f32 acc): Out = X[ROWS,C] * Wt^T, one wave / 16 rows.
// vlayout==0 -> Out f16 [B][T][H];  vlayout==1 -> Out f16 [B][H][T] (for V).
// ---------------------------------------------------------------------------
__global__ __launch_bounds__(32)
void proj_kernel(const float* __restrict__ X, const f16* __restrict__ Wt,
                 f16* __restrict__ Out, int vlayout) {
    const int r0   = blockIdx.x * 16;
    const int lane = threadIdx.x;
    const int m    = lane & 15;
    const int kh   = lane >> 4;

    v8f acc[8];
#pragma unroll
    for (int j = 0; j < 8; ++j) acc[j] = {};

    const float* arow = X + (size_t)(r0 + m) * C_DIM;

    for (int c0 = 0; c0 < C_DIM; c0 += 32) {
        // 16-bit A layout: lane holds K in [8*kh,8*kh+8) and [16+8*kh,+8)
        v4f a0 = *(const v4f*)(arow + c0 + 8 * kh);
        v4f a1 = *(const v4f*)(arow + c0 + 8 * kh + 4);
        v4f a2 = *(const v4f*)(arow + c0 + 16 + 8 * kh);
        v4f a3 = *(const v4f*)(arow + c0 + 16 + 8 * kh + 4);
        v16h a;
#pragma unroll
        for (int e = 0; e < 4; ++e) {
            a[e]      = (f16)a0[e];
            a[4 + e]  = (f16)a1[e];
            a[8 + e]  = (f16)a2[e];
            a[12 + e] = (f16)a3[e];
        }
#pragma unroll
        for (int j = 0; j < 8; ++j) {
            // B[k][n] = W[c0+k][16j+n] = Wt[16j+n][c0+k]  (contiguous in k)
            const f16* bp = Wt + (size_t)(16 * j + m) * C_DIM + c0;
            v8h b0 = *(const v8h*)(bp + 8 * kh);
            v8h b1 = *(const v8h*)(bp + 16 + 8 * kh);
            acc[j] = wmma16(a, cat8(b0, b1), acc[j]);
        }
    }

    // D-layout: vgpr i, lane -> row M = i + 8*kh, col N = m
#pragma unroll
    for (int j = 0; j < 8; ++j) {
#pragma unroll
        for (int i = 0; i < 8; ++i) {
            const int row = r0 + i + 8 * kh;       // = t*B + b
            const int t = row >> 3;
            const int b = row & (B_DIM - 1);
            const int h = 16 * j + m;
            const size_t o = vlayout
                ? ((size_t)b * H_DIM + h) * T_DIM + t     // [B][H][T]
                : ((size_t)b * T_DIM + t) * H_DIM + h;    // [B][T][H]
            Out[o] = (f16)acc[j][i];
        }
    }
}

// ---------------------------------------------------------------------------
// Flash attention, f16 WMMA / f32 acc, one wave per (batch, 16 query rows).
// K/V tiles staged into LDS by the Tensor Data Mover, double-buffered.
// Qh/Kh: f16 [B][T][H];  Vt: f16 [B][H][T];  Out: fp32 [T][B][O].
// ---------------------------------------------------------------------------
__global__ __launch_bounds__(32)
void attn_kernel(const f16* __restrict__ Qh, const f16* __restrict__ Kh,
                 const f16* __restrict__ Vt, float* __restrict__ Out) {
    __shared__ __align__(16) f16 kTile[2][32 * 128];   // [s=32][h=128] per buf
    __shared__ __align__(16) f16 vTile[2][128 * 32];   // [h=128][s=32] per buf
    __shared__ __align__(16) f16 pTile[16 * 32];       // P: 16 rows x 32 keys

    const int b    = blockIdx.y;
    const int t0   = blockIdx.x * 16;
    const int lane = threadIdx.x;
    const int m    = lane & 15;
    const int kh   = lane >> 4;

    const float scale = 0.08838834764831845f;   // 1/sqrt(128)

    const f16* kbat = Kh + (size_t)b * T_DIM * H_DIM;
    const f16* vbat = Vt + (size_t)b * H_DIM * T_DIM;
    const unsigned koff[2] = { (unsigned)(unsigned long long)&kTile[0][0],
                               (unsigned)(unsigned long long)&kTile[1][0] };
    const unsigned voff[2] = { (unsigned)(unsigned long long)&vTile[0][0],
                               (unsigned)(unsigned long long)&vTile[1][0] };

    // Q tile [16 x 128] as 4 A-fragments (32 VGPRs), kept in registers.
    v16h qa[4];
    const f16* qrow = Qh + ((size_t)b * T_DIM + t0 + m) * H_DIM;
#pragma unroll
    for (int kk = 0; kk < 4; ++kk) {
        v8h lo = *(const v8h*)(qrow + 32 * kk + 8 * kh);
        v8h hi = *(const v8h*)(qrow + 32 * kk + 16 + 8 * kh);
        qa[kk] = cat8(lo, hi);
    }

    v8f ctx[8];
#pragma unroll
    for (int j = 0; j < 8; ++j) ctx[j] = {};
    float mrow[8], lrow[8];
#pragma unroll
    for (int i = 0; i < 8; ++i) { mrow[i] = -3.0e38f; lrow[i] = 0.0f; }

    const int nTiles = (t0 >> 5) + 1;    // 32-key tiles covering s <= t0+15

    // Prologue: TDM-fetch tile 0 into buffer 0 (2 tensor ops in flight).
    tdm_load_2d_f16(koff[0], kbat, /*w=*/H_DIM, /*h=*/32, /*stride=*/H_DIM);
    tdm_load_2d_f16(voff[0], vbat, /*w=*/32, /*h=*/H_DIM, /*stride=*/T_DIM);

    for (int it = 0; it < nTiles; ++it) {
        const int s0  = it * 32;
        const int buf = it & 1;

        if (it + 1 < nTiles) {          // prefetch next tile into other buffer
            const int s1 = s0 + 32;
            tdm_load_2d_f16(koff[buf ^ 1], kbat + (size_t)s1 * H_DIM,
                            H_DIM, 32, H_DIM);
            tdm_load_2d_f16(voff[buf ^ 1], vbat + s1, 32, H_DIM, T_DIM);
            __builtin_amdgcn_s_wait_tensorcnt(2);   // current tile complete
        } else {
            __builtin_amdgcn_s_wait_tensorcnt(0);
        }
        asm volatile("" ::: "memory");   // TDM wrote LDS behind the compiler

        // ---- S = Q * K^T : 2 n-halves x 4 k-steps, K from LDS ----
        v8f sacc[2];
        sacc[0] = {};
        sacc[1] = {};
#pragma unroll
        for (int nh = 0; nh < 2; ++nh) {
            const f16* kr = &kTile[buf][(16 * nh + m) * 128];
#pragma unroll
            for (int kk = 0; kk < 4; ++kk) {
                v8h lo = *(const v8h*)(kr + 32 * kk + 8 * kh);
                v8h hi = *(const v8h*)(kr + 32 * kk + 16 + 8 * kh);
                sacc[nh] = wmma16(qa[kk], cat8(lo, hi), sacc[nh]);
            }
        }

        // ---- scale + causal mask + online softmax (per D-layout row) ----
        __syncthreads();   // pTile WAR vs previous iteration's loads
#pragma unroll
        for (int i = 0; i < 8; ++i) {
            const int M = i + 8 * kh;
            const int t = t0 + M;
            float v0 = sacc[0][i] * scale;
            float v1 = sacc[1][i] * scale;
            v0 = (s0 + m > t)      ? v0 - 1.0e9f : v0;
            v1 = (s0 + 16 + m > t) ? v1 - 1.0e9f : v1;

            float r = fmaxf(v0, v1);
            r = fmaxf(r, __shfl_xor(r, 1, 32));
            r = fmaxf(r, __shfl_xor(r, 2, 32));
            r = fmaxf(r, __shfl_xor(r, 4, 32));
            r = fmaxf(r, __shfl_xor(r, 8, 32));   // row max within 16-lane half

            const float mnew = fmaxf(mrow[i], r);
            const float corr = __expf(mrow[i] - mnew);
            const float p0   = __expf(v0 - mnew);
            const float p1   = __expf(v1 - mnew);

            float ps = p0 + p1;
            ps += __shfl_xor(ps, 1, 32);
            ps += __shfl_xor(ps, 2, 32);
            ps += __shfl_xor(ps, 4, 32);
            ps += __shfl_xor(ps, 8, 32);          // row sum

            lrow[i] = lrow[i] * corr + ps;
            mrow[i] = mnew;
#pragma unroll
            for (int j = 0; j < 8; ++j) ctx[j][i] *= corr;

            pTile[M * 32 + m]      = (f16)p0;     // D-layout -> row-major LDS
            pTile[M * 32 + 16 + m] = (f16)p1;
        }
        __syncthreads();

        // ---- P back in A-layout: two b128 LDS loads ----
        v8h pl0 = *(const v8h*)(&pTile[m * 32 + 8 * kh]);
        v8h pl1 = *(const v8h*)(&pTile[m * 32 + 16 + 8 * kh]);
        const v16h pa = cat8(pl0, pl1);

        // ---- ctx += P(16x32) * V(32x16), V from LDS ----
#pragma unroll
        for (int j = 0; j < 8; ++j) {
            const f16* vr = &vTile[buf][(16 * j + m) * 32];
            v8h lo = *(const v8h*)(vr + 8 * kh);
            v8h hi = *(const v8h*)(vr + 16 + 8 * kh);
            ctx[j] = wmma16(pa, cat8(lo, hi), ctx[j]);
        }
    }

    // ---- epilogue: normalize and write fp32 [T,B,O] ----
#pragma unroll
    for (int j = 0; j < 8; ++j) {
#pragma unroll
        for (int i = 0; i < 8; ++i) {
            const int t = t0 + i + 8 * kh;
            Out[((size_t)t * B_DIM + b) * H_DIM + 16 * j + m] = ctx[j][i] / lrow[i];
        }
    }
}

extern "C" void kernel_launch(void* const* d_in, const int* in_sizes, int n_in,
                              void* d_out, int out_size, void* d_ws, size_t ws_size,
                              hipStream_t stream) {
    const float* q  = (const float*)d_in[0];
    const float* k  = (const float*)d_in[1];
    const float* v  = (const float*)d_in[2];
    // d_in[3] = additive causal mask: applied analytically in attn_kernel
    const float* Wq = (const float*)d_in[4];
    const float* Wk = (const float*)d_in[5];
    const float* Wv = (const float*)d_in[6];
    float* out = (float*)d_out;

    // Workspace layout (f16): Qh/Kh [B][T][H], Vt [B][H][T], 3x transposed weights
    f16* Qh  = (f16*)d_ws;                               // 4 MB
    f16* Kh  = Qh + (size_t)ROWS * H_DIM;                // 4 MB
    f16* Vt  = Kh + (size_t)ROWS * H_DIM;                // 4 MB
    f16* Wtq = Vt + (size_t)ROWS * H_DIM;                // 256 KB each
    f16* Wtk = Wtq + (size_t)C_DIM * H_DIM;
    f16* Wtv = Wtk + (size_t)C_DIM * H_DIM;

    const int wgrid = (C_DIM * H_DIM) / 256;             // 512
    wtrans_kernel<<<dim3(wgrid), dim3(256), 0, stream>>>(Wq, Wtq);
    wtrans_kernel<<<dim3(wgrid), dim3(256), 0, stream>>>(Wk, Wtk);
    wtrans_kernel<<<dim3(wgrid), dim3(256), 0, stream>>>(Wv, Wtv);

    proj_kernel<<<dim3(ROWS / 16), dim3(32), 0, stream>>>(q, Wtq, Qh, 0);
    proj_kernel<<<dim3(ROWS / 16), dim3(32), 0, stream>>>(k, Wtk, Kh, 0);
    proj_kernel<<<dim3(ROWS / 16), dim3(32), 0, stream>>>(v, Wtv, Vt, 1);

    attn_kernel<<<dim3(T_DIM / 16, B_DIM), dim3(32), 0, stream>>>(Qh, Kh, Vt, out);
}
